// InstructionMapper_89103391523485
// MI455X (gfx1250) — matrix operand order repaired and checked by reading, not compiled, not run on hardware
//
#include <hip/hip_runtime.h>
#include <hip/hip_bf16.h>

#define B_ROWS 32768
#define DD 448
#define CC 896

typedef __attribute__((ext_vector_type(16))) _Float16 v16h;
typedef __attribute__((ext_vector_type(8)))  _Float16 v8h;
typedef __attribute__((ext_vector_type(8)))  float    v8f;
typedef __attribute__((ext_vector_type(4)))  float    v4f;

enum { EPI_NONE = 0, EPI_SILU = 1, EPI_ADD = 2 };

__device__ __forceinline__ float fast_silu(float y) {
    return y * __builtin_amdgcn_rcpf(1.0f + __expf(-y));
}

// ---------------------------------------------------------------------------
// WMMA GEMM: Y[B,N] = epilogue( X[B,K](f32->f16) @ W[K,N](f32->f16) + bias )
// 128x64 tile / WG, 8 waves; wave w owns rows [w*16, w*16+16) x all 4 col tiles.
// LDS double-buffered: next tile's global loads are in flight under the WMMAs,
// one barrier per k-step.
// ---------------------------------------------------------------------------
template<int EPI>
__global__ __launch_bounds__(256) void gemm_wmma(
    const float* __restrict__ X, int ldx,
    const float* __restrict__ W, int ldw,
    const float* __restrict__ bias,
    const float* __restrict__ R, int ldr,
    float* __restrict__ Y, int ldy,
    int K)
{
    constexpr int BM = 128, BN = 64, BK = 32;
    constexpr int LDA = 40;   // halves; row pitch 80B -> 16B-aligned b128 frags
    constexpr int LDB = 40;

    __shared__ __align__(16) _Float16 lsA[2][BM * LDA];   // [m][k]
    __shared__ __align__(16) _Float16 lsB[2][BN * LDB];   // [n][k] (transposed)

    const int tid  = threadIdx.x;
    const int lane = tid & 31;
    const int wave = tid >> 5;          // m-tile index 0..7
    const int row0 = blockIdx.y * BM;
    const int col0 = blockIdx.x * BN;
    const int m    = lane & 15;
    const int hi   = lane >> 4;

    // staging coordinates
    const int ar = tid >> 1;            // A row 0..127
    const int ac = (tid & 1) * 16;      // A col base 0 or 16
    const int bn = tid & 63;            // B n 0..63
    const int bk = (tid >> 6) * 8;      // B k base 0,8,16,24

    const float* Xp = X + (size_t)(row0 + ar) * ldx + ac;   // + kt
    const float* Wp = W + (size_t)bk * ldw + (col0 + bn);   // + kt*ldw (+ j*ldw)

    // next-tile staging registers
    v4f  a0, a1, a2, a3;
    float wr[8];

    auto GLOAD = [&](int kt) {
        const v4f* s4 = (const v4f*)(Xp + kt);
        a0 = s4[0]; a1 = s4[1]; a2 = s4[2]; a3 = s4[3];
        const float* wp = Wp + (size_t)kt * ldw;
        #pragma unroll
        for (int j = 0; j < 8; ++j) wr[j] = wp[(size_t)j * ldw];
    };
    auto SSTORE = [&](int p) {
        v8h h0, h1, hb;
        #pragma unroll
        for (int j = 0; j < 4; ++j) {
            h0[j]     = (_Float16)a0[j];
            h0[4 + j] = (_Float16)a1[j];
            h1[j]     = (_Float16)a2[j];
            h1[4 + j] = (_Float16)a3[j];
        }
        #pragma unroll
        for (int j = 0; j < 8; ++j) hb[j] = (_Float16)wr[j];
        *(v8h*)&lsA[p][ar * LDA + ac]     = h0;
        *(v8h*)&lsA[p][ar * LDA + ac + 8] = h1;
        *(v8h*)&lsB[p][bn * LDB + bk]     = hb;
    };

    v8f acc[4] = {v8f{}, v8f{}, v8f{}, v8f{}};

    GLOAD(0);
    SSTORE(0);
    __syncthreads();

    int p = 0;
    for (int kt = 0; kt < K; kt += BK) {
        const bool more = (kt + BK) < K;
        // global loads for next tile go in flight under this tile's WMMAs
        if (more) GLOAD(kt + BK);
        // pre-warm L2 two tiles ahead (global_prefetch_b8)
        if (kt + 2 * BK < K) {
            __builtin_prefetch(Xp + kt + 2 * BK, 0, 1);
            __builtin_prefetch(Wp + (size_t)(kt + 2 * BK) * ldw, 0, 1);
        }

        // ---- all fragment loads first, then 4 WMMAs ----
        union { v16h v; v8h h[2]; } ua, ub0, ub1, ub2, ub3;
        const _Float16* pa = &lsA[p][(wave * 16 + m) * LDA + hi * 16];
        ua.h[0] = *(const v8h*)(pa);
        ua.h[1] = *(const v8h*)(pa + 8);
        const _Float16* pb = &lsB[p][m * LDB + hi * 16];
        ub0.h[0] = *(const v8h*)(pb);
        ub0.h[1] = *(const v8h*)(pb + 8);
        ub1.h[0] = *(const v8h*)(pb + 16 * LDB);
        ub1.h[1] = *(const v8h*)(pb + 16 * LDB + 8);
        ub2.h[0] = *(const v8h*)(pb + 32 * LDB);
        ub2.h[1] = *(const v8h*)(pb + 32 * LDB + 8);
        ub3.h[0] = *(const v8h*)(pb + 48 * LDB);
        ub3.h[1] = *(const v8h*)(pb + 48 * LDB + 8);

        acc[0] = __builtin_amdgcn_wmma_f32_16x16x32_f16(false, ua.v, false, ub0.v,
                                                        (short)0, acc[0], false, false);
        acc[1] = __builtin_amdgcn_wmma_f32_16x16x32_f16(false, ua.v, false, ub1.v,
                                                        (short)0, acc[1], false, false);
        acc[2] = __builtin_amdgcn_wmma_f32_16x16x32_f16(false, ua.v, false, ub2.v,
                                                        (short)0, acc[2], false, false);
        acc[3] = __builtin_amdgcn_wmma_f32_16x16x32_f16(false, ua.v, false, ub3.v,
                                                        (short)0, acc[3], false, false);

        if (more) {
            SSTORE(p ^ 1);
            __syncthreads();
        }
        p ^= 1;
    }

    // ---- fused epilogue ----
    const int orow = row0 + wave * 16 + hi * 8;   // + v gives the row
    #pragma unroll
    for (int nt = 0; nt < 4; ++nt) {
        const int ocol = col0 + nt * 16 + m;
        const float bb = bias[ocol];
        #pragma unroll
        for (int v = 0; v < 8; ++v) {
            float y = acc[nt][v] + bb;
            if (EPI == EPI_SILU) y = fast_silu(y);
            if (EPI == EPI_ADD)  y += R[(size_t)(orow + v) * ldr + ocol];
            Y[(size_t)(orow + v) * ldy + ocol] = y;
        }
    }
}

// ---------------------------------------------------------------------------
// AdaLN: xn = LN(h) * (1 + ss[:, :448]) + ss[:, 448:]   (one wave32 per row)
// ---------------------------------------------------------------------------
__global__ __launch_bounds__(256) void adaln_kernel(
    const float* __restrict__ h, const float* __restrict__ ss,
    float* __restrict__ xn)
{
    const int lane = threadIdx.x & 31;
    const int row  = blockIdx.x * 8 + (threadIdx.x >> 5);
    const float* hr = h + (size_t)row * DD;
    float vals[14];
    float s = 0.f;
    #pragma unroll
    for (int j = 0; j < 14; ++j) { vals[j] = hr[lane + j * 32]; s += vals[j]; }
    #pragma unroll
    for (int o = 16; o > 0; o >>= 1) s += __shfl_xor(s, o, 32);
    const float mean = s * (1.0f / DD);
    float var = 0.f;
    #pragma unroll
    for (int j = 0; j < 14; ++j) { float d = vals[j] - mean; var += d * d; }
    #pragma unroll
    for (int o = 16; o > 0; o >>= 1) var += __shfl_xor(var, o, 32);
    const float inv = __builtin_amdgcn_rsqf(var * (1.0f / DD) + 1e-5f);
    const float* sr = ss + (size_t)row * CC;
    float* xr = xn + (size_t)row * DD;
    #pragma unroll
    for (int j = 0; j < 14; ++j) {
        const int c = lane + j * 32;
        xr[c] = (vals[j] - mean) * inv * (1.0f + sr[c]) + sr[DD + c];
    }
}

// Final affine LayerNorm: xn = LN(h) * g + b
__global__ __launch_bounds__(256) void ln_affine_kernel(
    const float* __restrict__ h, const float* __restrict__ g,
    const float* __restrict__ b, float* __restrict__ xn)
{
    const int lane = threadIdx.x & 31;
    const int row  = blockIdx.x * 8 + (threadIdx.x >> 5);
    const float* hr = h + (size_t)row * DD;
    float vals[14];
    float s = 0.f;
    #pragma unroll
    for (int j = 0; j < 14; ++j) { vals[j] = hr[lane + j * 32]; s += vals[j]; }
    #pragma unroll
    for (int o = 16; o > 0; o >>= 1) s += __shfl_xor(s, o, 32);
    const float mean = s * (1.0f / DD);
    float var = 0.f;
    #pragma unroll
    for (int j = 0; j < 14; ++j) { float d = vals[j] - mean; var += d * d; }
    #pragma unroll
    for (int o = 16; o > 0; o >>= 1) var += __shfl_xor(var, o, 32);
    const float inv = __builtin_amdgcn_rsqf(var * (1.0f / DD) + 1e-5f);
    float* xr = xn + (size_t)row * DD;
    #pragma unroll
    for (int j = 0; j < 14; ++j) {
        const int c = lane + j * 32;
        xr[c] = (vals[j] - mean) * inv * g[c] + b[c];
    }
}

// Sinusoidal time embedding: te0[b, i] = sin/cos(t[b] * freq_i)
__global__ __launch_bounds__(256) void sinus_kernel(
    const float* __restrict__ t, float* __restrict__ te0)
{
    const int idx = blockIdx.x * 256 + threadIdx.x;   // over B*448
    const int row = idx / DD;
    const int c   = idx - row * DD;
    const float tv = t[row];
    const int   i  = (c < 224) ? c : (c - 224);
    const float f  = __expf((float)i * (-9.210340371976184f / 223.0f));
    const float a  = tv * f;
    te0[idx] = (c < 224) ? __sinf(a) : __cosf(a);
}

// ---------------------------------------------------------------------------
extern "C" void kernel_launch(void* const* d_in, const int* in_sizes, int n_in,
                              void* d_out, int out_size, void* d_ws, size_t ws_size,
                              hipStream_t stream)
{
    (void)in_sizes; (void)n_in; (void)out_size; (void)ws_size;
    const float* x     = (const float*)d_in[0];
    const float* t     = (const float*)d_in[1];
    const float* style = (const float*)d_in[2];
    const float* tw1   = (const float*)d_in[3];
    const float* tb1   = (const float*)d_in[4];
    const float* tw2   = (const float*)d_in[5];
    const float* tb2   = (const float*)d_in[6];
    const float* cw    = (const float*)d_in[7];
    const float* cb    = (const float*)d_in[8];
    const float* iw    = (const float*)d_in[9];
    const float* ib    = (const float*)d_in[10];
    const float* adaw  = (const float*)d_in[11];
    const float* adab  = (const float*)d_in[12];
    const float* l1w   = (const float*)d_in[13];
    const float* l1b   = (const float*)d_in[14];
    const float* l2w   = (const float*)d_in[15];
    const float* l2b   = (const float*)d_in[16];
    const float* fng   = (const float*)d_in[17];
    const float* fnb   = (const float*)d_in[18];
    const float* ow    = (const float*)d_in[19];
    const float* ob    = (const float*)d_in[20];
    float* out = (float*)d_out;

    // workspace layout (floats)
    float* ws   = (float*)d_ws;
    float* cond = ws;                                   // [B, 896]
    float* h    = cond + (size_t)B_ROWS * CC;           // [B, 448]
    float* ss   = h    + (size_t)B_ROWS * DD;           // [B, 896]
    float* xn   = ss   + (size_t)B_ROWS * CC;           // [B, 448]
    float* hid  = xn   + (size_t)B_ROWS * DD;           // [B, 896]
    float* te0  = ss;    // aliased (used only before the block loop)
    float* te1  = hid;   // aliased

    const dim3 blk(256);
    const dim3 gN448(DD / 64, B_ROWS / 128);
    const dim3 gN896(CC / 64, B_ROWS / 128);

    // time embedding
    sinus_kernel<<<(B_ROWS * DD) / 256, blk, 0, stream>>>(t, te0);
    // te1 = silu(te0 @ tw1 + tb1)
    gemm_wmma<EPI_SILU><<<gN448, blk, 0, stream>>>(te0, DD, tw1, DD, tb1,
                                                   nullptr, 0, te1, DD, DD);
    // cond[:, :448] = te1 @ tw2 + tb2
    gemm_wmma<EPI_NONE><<<gN448, blk, 0, stream>>>(te1, DD, tw2, DD, tb2,
                                                   nullptr, 0, cond, CC, DD);
    // cond[:, 448:] = style @ cw + cb
    gemm_wmma<EPI_NONE><<<gN448, blk, 0, stream>>>(style, 1024, cw, DD, cb,
                                                   nullptr, 0, cond + DD, CC, 1024);
    // h = x @ iw + ib
    gemm_wmma<EPI_NONE><<<gN448, blk, 0, stream>>>(x, DD, iw, DD, ib,
                                                   nullptr, 0, h, DD, DD);

    for (int d = 0; d < 6; ++d) {
        // ss = cond @ ada_w[d] + ada_b[d]
        gemm_wmma<EPI_NONE><<<gN896, blk, 0, stream>>>(
            cond, CC, adaw + (size_t)d * CC * CC, CC, adab + (size_t)d * CC,
            nullptr, 0, ss, CC, CC);
        // xn = LN(h) * (1+scale) + shift
        adaln_kernel<<<B_ROWS / 8, blk, 0, stream>>>(h, ss, xn);
        // hid = silu(xn @ lin1_w[d] + lin1_b[d])
        gemm_wmma<EPI_SILU><<<gN896, blk, 0, stream>>>(
            xn, DD, l1w + (size_t)d * DD * CC, CC, l1b + (size_t)d * CC,
            nullptr, 0, hid, CC, DD);
        // h = h + hid @ lin2_w[d] + lin2_b[d]
        gemm_wmma<EPI_ADD><<<gN448, blk, 0, stream>>>(
            hid, CC, l2w + (size_t)d * CC * DD, DD, l2b + (size_t)d * DD,
            h, DD, h, DD, CC);
    }

    // final LN + output projection
    ln_affine_kernel<<<B_ROWS / 8, blk, 0, stream>>>(h, fng, fnb, xn);
    gemm_wmma<EPI_NONE><<<gN448, blk, 0, stream>>>(xn, DD, ow, DD, ob,
                                                   nullptr, 0, out, DD, DD);
}
